// LSTM_13975823581657
// MI455X (gfx1250) — matrix operand order repaired and checked by reading, not compiled, not run on hardware
//
#include <hip/hip_runtime.h>
#include <hip/hip_bf16.h>
#include <cstdint>
#include <cstddef>

// ---------------------------------------------------------------------------
// 2-layer LSTM for MI455X (gfx1250, wave32, WMMA).
//
//   per layer l:
//     G = input_seq @ W_i[l]     (big WMMA GEMM [32768,1024]x[1024,4096], LDS-tiled,
//                                 async global->LDS weight staging when available)
//     for t in 0..511 (serial):  gates = G[t] + b + h_prev @ W_h[l]
//         step kernel is LDS-free / barrier-free: A (h in bf16 hi/lo ping-pong
//         planes) and B (pre-transposed bf16 weights, L2-resident) are fragment-
//         loaded straight from global into WMMA layouts.
//
// Precision: bf16x3 split (Ahi*Bhi + Alo*Bhi + Ahi*Blo, fp32 accum) ~ fp32.
// ---------------------------------------------------------------------------

typedef __attribute__((ext_vector_type(16))) __bf16 v16bf;
typedef __attribute__((ext_vector_type(8)))  __bf16 v8bf;
typedef __attribute__((ext_vector_type(8)))  float  v8f;
typedef int v4i __attribute__((vector_size(4 * sizeof(int))));

#define LAYERS 2
#define BB     64
#define LL     512
#define KDIM   1024        // IN_SIZE == HID
#define HIDN   1024
#define NG     4096        // 4*HID
#define MTOT   (BB * LL)   // 32768

#if __has_builtin(__builtin_amdgcn_global_load_async_to_lds_b128) && \
    __has_builtin(__builtin_amdgcn_s_wait_asynccnt)
#define USE_ASYNC_LDS 1
#define AS_GLOBAL __attribute__((address_space(1)))
#define AS_LDS    __attribute__((address_space(3)))
#else
#define USE_ASYNC_LDS 0
#endif

__device__ __forceinline__ float sigmf(float x) { return 1.0f / (1.0f + __expf(-x)); }

__device__ __forceinline__ v8f wmma_bf16(v16bf a, v16bf b, v8f c) {
  // (neg_a, A, neg_b, B, c_mod, C, reuse_a, reuse_b)
  return __builtin_amdgcn_wmma_f32_16x16x32_bf16(false, a, false, b, (short)0, c,
                                                 false, false);
}

// A-matrix 16x32 bf16 fragment, K contiguous (matches B):
//   lanes 0-15 : M=lane,    K=0..15
//   lanes 16-31: M=lane-16, K=16..31
// rowp points at element K=0 of this lane's row; half = lane>>4.
__device__ __forceinline__ v16bf load_afrag(const __bf16* rowp, int half) {
  union { v16bf v; v8bf h[2]; } u;
  u.h[0] = *(const v8bf*)(rowp + half * 16);
  u.h[1] = *(const v8bf*)(rowp + half * 16 + 8);
  return u.v;
}

// ---------------------------------------------------------------------------
// Weight transpose + fp32 -> (bf16_hi, bf16_lo) split.
// In : W[l][k][n] fp32.   Out: T*[l][n][k] bf16 (hi and lo planes).
// ---------------------------------------------------------------------------
__global__ __launch_bounds__(256) void wt_transpose(
    const float* __restrict__ Wi, const float* __restrict__ Wh,
    __bf16* __restrict__ Tih, __bf16* __restrict__ Til,
    __bf16* __restrict__ Thh, __bf16* __restrict__ Thl) {
  size_t idx = (size_t)blockIdx.x * 256 + threadIdx.x;   // over [L][NG][K], k fastest
  int k = (int)(idx & (KDIM - 1));
  size_t r = idx >> 10;
  int n = (int)(r & (NG - 1));
  int l = (int)(r >> 12);
  size_t src = ((size_t)l * KDIM + k) * NG + n;
  float wi = Wi[src];
  __bf16 wih = (__bf16)wi;
  Tih[idx] = wih;
  Til[idx] = (__bf16)(wi - (float)wih);
  float wh = Wh[src];
  __bf16 whh = (__bf16)wh;
  Thh[idx] = whh;
  Thl[idx] = (__bf16)(wh - (float)whh);
}

// ---------------------------------------------------------------------------
// h_0 fp32 -> bf16 hi/lo planes (one-time, both layers).
// ---------------------------------------------------------------------------
__global__ __launch_bounds__(256) void h0_convert(
    const float* __restrict__ h0, __bf16* __restrict__ hh, __bf16* __restrict__ hl) {
  size_t i = (size_t)blockIdx.x * 256 + threadIdx.x;  // over LAYERS*BB*HIDN
  float f = h0[i];
  __bf16 h = (__bf16)f;
  hh[i] = h;
  hl[i] = (__bf16)(f - (float)h);
}

// ---------------------------------------------------------------------------
// Big input GEMM: G[m][n] = sum_k A[m][k] * Wt[n][k],  m = t*64 + b.
// a_bt_layout==1: A row m lives at x[(b*512 + t)*1024]  (x is [B,L,K]).
// Block: 256 thr (8 waves), tile M=64 x N=64; K-chunk 32 via LDS.
// Weight tiles staged with GLOBAL_LOAD_ASYNC_TO_LDS when the builtin exists.
// ---------------------------------------------------------------------------
__global__ __launch_bounds__(256) void gemm_gates(
    const float* __restrict__ A,
    const __bf16* __restrict__ Wth, const __bf16* __restrict__ Wtl,
    float* __restrict__ G, int a_bt_layout) {
  __shared__ __bf16 Ash[64][32];
  __shared__ __bf16 Asl[64][32];
  __shared__ __bf16 Bsh[64][32];
  __shared__ __bf16 Bsl[64][32];

  const int tid = threadIdx.x;
  const int j0 = blockIdx.x * 64;
  const int m0 = blockIdx.y * 64;

  const int wid = tid >> 5, lane = tid & 31;
  const int mt = wid & 3;
  const int nb = (wid >> 2) * 32;
  const int half = lane >> 4, lrow = lane & 15;

  v8f acc0 = {}; v8f acc1 = {};

  const int arow = tid >> 2;
  const int aseg = (tid & 3) * 8;
  const int am = m0 + arow;
  size_t abase;
  if (a_bt_layout) {
    int bidx = am & (BB - 1);
    int tt = am >> 6;
    abase = ((size_t)bidx * LL + tt) * KDIM;
  } else {
    abase = (size_t)am * KDIM;
  }
  const size_t boff = (size_t)(j0 + arow) * KDIM + aseg;

  for (int kk = 0; kk < KDIM; kk += 32) {
    {  // B tiles (pre-transposed bf16, contiguous in k): global -> LDS
#if USE_ASYNC_LDS
      // builtin signature (from clang diagnostic): param0 = int4 AS(1)*, 16B payload
      __builtin_amdgcn_global_load_async_to_lds_b128(
          (AS_GLOBAL v4i*)(Wth + boff + kk),
          (AS_LDS v4i*)&Bsh[arow][aseg], 0, 0);
      __builtin_amdgcn_global_load_async_to_lds_b128(
          (AS_GLOBAL v4i*)(Wtl + boff + kk),
          (AS_LDS v4i*)&Bsl[arow][aseg], 0, 0);
#else
      const v8bf* sh = (const v8bf*)(Wth + boff + kk);
      const v8bf* sl = (const v8bf*)(Wtl + boff + kk);
      *(v8bf*)&Bsh[arow][aseg] = *sh;
      *(v8bf*)&Bsl[arow][aseg] = *sl;
      if (kk + 32 < KDIM) {
        __builtin_prefetch((const char*)sh + 64, 0, 0);
        __builtin_prefetch((const char*)sl + 64, 0, 0);
      }
#endif
    }
    {  // A tile: fp32 -> hi/lo bf16 (needs VALU conversion, stays sync)
      const float* s = A + abase + kk + aseg;
      __bf16* dh = &Ash[arow][aseg];
      __bf16* dl = &Asl[arow][aseg];
#pragma unroll
      for (int i = 0; i < 8; ++i) {
        float f = s[i];
        __bf16 h = (__bf16)f;
        dh[i] = h;
        dl[i] = (__bf16)(f - (float)h);
      }
    }
#if USE_ASYNC_LDS
    __builtin_amdgcn_s_wait_asynccnt(0);
#endif
    __syncthreads();

    v16bf ah = load_afrag(&Ash[mt * 16 + lrow][0], half);
    v16bf al = load_afrag(&Asl[mt * 16 + lrow][0], half);
    // B 32x16 fragment: lane = N column, K contiguous (half selects K 0-15/16-31)
    v16bf bh0 = *(const v16bf*)&Bsh[nb + lrow][half * 16];
    v16bf bl0 = *(const v16bf*)&Bsl[nb + lrow][half * 16];
    v16bf bh1 = *(const v16bf*)&Bsh[nb + 16 + lrow][half * 16];
    v16bf bl1 = *(const v16bf*)&Bsl[nb + 16 + lrow][half * 16];

    acc0 = wmma_bf16(ah, bh0, acc0);
    acc0 = wmma_bf16(al, bh0, acc0);
    acc0 = wmma_bf16(ah, bl0, acc0);
    acc1 = wmma_bf16(ah, bh1, acc1);
    acc1 = wmma_bf16(al, bh1, acc1);
    acc1 = wmma_bf16(ah, bl1, acc1);
    __syncthreads();
  }

#pragma unroll
  for (int v = 0; v < 8; ++v) {
    int m = m0 + mt * 16 + half * 8 + v;  // C/D layout: M = v + 8*(lane>>4)
    int n = j0 + nb + lrow;
    G[(size_t)m * NG + n] = acc0[v];
    G[(size_t)m * NG + n + 16] = acc1[v];
  }
}

// ---------------------------------------------------------------------------
// One recurrent step, all 4 gates fused.  LDS-free, barrier-free:
// A fragments come from bf16 hi/lo h-planes (256KB, cache-resident),
// B fragments from pre-transposed bf16 weights (16MB/layer, L2-resident).
// Block owns hidden slice j0..j0+31; wave (mt,nt) computes a 16x16 tile per gate.
// Grid = HID/32 = 32 blocks x 8 waves.
// ---------------------------------------------------------------------------
__global__ __launch_bounds__(256) void lstm_step(
    const float* __restrict__ G,
    const __bf16* __restrict__ Wth, const __bf16* __restrict__ Wtl,
    const float* __restrict__ bi, const float* __restrict__ bh,
    const __bf16* __restrict__ hph, const __bf16* __restrict__ hpl,  // h_prev hi/lo
    __bf16* __restrict__ nxh, __bf16* __restrict__ nxl,              // h_next hi/lo
    float* __restrict__ hstate, float* __restrict__ c_state,
    float* __restrict__ hout, int t, int out_bt) {
  const int tid = threadIdx.x;
  const int j0 = blockIdx.x * 32;

  const int wid = tid >> 5, lane = tid & 31;
  const int mt = wid & 3;       // batch tile 0..3  (M = 64 = full batch)
  const int nt = wid >> 2;      // 0..1 within the 32-wide hidden slice
  const int half = lane >> 4, lrow = lane & 15;

  v8f acc[4] = {{}, {}, {}, {}};

  // per-lane fragment bases
  const size_t arow = (size_t)(mt * 16 + lrow) * KDIM;       // A row (batch)
  const int jn = j0 + nt * 16 + lrow;                        // B column within gate
  size_t bb[4];
#pragma unroll
  for (int g = 0; g < 4; ++g)
    bb[g] = (size_t)(g * HIDN + jn) * KDIM + (size_t)half * 16;

  for (int kk = 0; kk < KDIM; kk += 32) {
    v16bf ah, al;
    {
      const __bf16* ph = hph + arow + kk + half * 16;
      const __bf16* pl = hpl + arow + kk + half * 16;
      union { v16bf v; v8bf h[2]; } uh, ul;
      uh.h[0] = *(const v8bf*)ph;
      uh.h[1] = *(const v8bf*)(ph + 8);
      ul.h[0] = *(const v8bf*)pl;
      ul.h[1] = *(const v8bf*)(pl + 8);
      ah = uh.v;
      al = ul.v;
    }
#pragma unroll
    for (int g = 0; g < 4; ++g) {
      v16bf bhf = *(const v16bf*)(Wth + bb[g] + kk);   // 32B, K-contiguous
      v16bf blf = *(const v16bf*)(Wtl + bb[g] + kk);
      acc[g] = wmma_bf16(ah, bhf, acc[g]);
      acc[g] = wmma_bf16(al, bhf, acc[g]);
      acc[g] = wmma_bf16(ah, blf, acc[g]);
      if (kk + 32 < KDIM)
        __builtin_prefetch((const char*)(Wth + bb[g] + kk) + 64, 0, 0);
    }
  }

  // epilogue: gates -> cell update.  Each (b, jj) owned by exactly one lane/v.
  const int jj = jn;
  const float bs0 = bi[jj]            + bh[jj];
  const float bs1 = bi[HIDN + jj]     + bh[HIDN + jj];
  const float bs2 = bi[2 * HIDN + jj] + bh[2 * HIDN + jj];
  const float bs3 = bi[3 * HIDN + jj] + bh[3 * HIDN + jj];

#pragma unroll
  for (int v = 0; v < 8; ++v) {
    const int b = mt * 16 + half * 8 + v;
    const size_t gm = ((size_t)t * BB + b) * NG;
    float gi = acc[0][v] + G[gm + jj]            + bs0;
    float gf = acc[1][v] + G[gm + HIDN + jj]     + bs1;
    float gz = acc[2][v] + G[gm + 2 * HIDN + jj] + bs2;
    float go = acc[3][v] + G[gm + 3 * HIDN + jj] + bs3;
    const size_t sidx = (size_t)b * HIDN + jj;
    float cn = sigmf(gf) * c_state[sidx] + sigmf(gi) * tanhf(gz);
    float hn = sigmf(go) * tanhf(cn);
    c_state[sidx] = cn;
    __bf16 hh = (__bf16)hn;                 // next-step A operand, pre-split
    nxh[sidx] = hh;
    nxl[sidx] = (__bf16)(hn - (float)hh);
    hstate[sidx] = hn;                      // fp32 mirror -> h_n output
    size_t oidx = out_bt ? ((size_t)b * LL + t) * HIDN + jj
                         : ((size_t)t * BB + b) * HIDN + jj;
    hout[oidx] = hn;
  }
}

// ---------------------------------------------------------------------------
extern "C" void kernel_launch(void* const* d_in, const int* in_sizes, int n_in,
                              void* d_out, int out_size, void* d_ws, size_t ws_size,
                              hipStream_t stream) {
  const float* x   = (const float*)d_in[0];
  const float* W_i = (const float*)d_in[1];
  const float* b_i = (const float*)d_in[2];
  const float* W_h = (const float*)d_in[3];
  const float* b_h = (const float*)d_in[4];
  const float* h0  = (const float*)d_in[5];
  const float* c0  = (const float*)d_in[6];
  float* out = (float*)d_out;

  // workspace carve-up (~706 MiB; MI455X has 432 GiB)
  char* p = (char*)d_ws;
  auto take = [&](size_t bytes) {
    char* r = p;
    p += (bytes + 255) & ~(size_t)255;
    return r;
  };
  const size_t WSZ = (size_t)LAYERS * NG * KDIM;  // per weight plane
  const size_t HS  = (size_t)LAYERS * BB * HIDN;  // per h/c state plane
  __bf16* Wti_h = (__bf16*)take(WSZ * sizeof(__bf16));
  __bf16* Wti_l = (__bf16*)take(WSZ * sizeof(__bf16));
  __bf16* Wth_h = (__bf16*)take(WSZ * sizeof(__bf16));
  __bf16* Wth_l = (__bf16*)take(WSZ * sizeof(__bf16));
  float* G      = (float*)take((size_t)MTOT * NG * sizeof(float));    // 512 MiB
  float* h0_all = (float*)take((size_t)MTOT * HIDN * sizeof(float));  // 128 MiB
  __bf16* hbA_h = (__bf16*)take(HS * sizeof(__bf16));  // h ping, hi
  __bf16* hbA_l = (__bf16*)take(HS * sizeof(__bf16));  // h ping, lo
  __bf16* hbB_h = (__bf16*)take(HS * sizeof(__bf16));  // h pong, hi
  __bf16* hbB_l = (__bf16*)take(HS * sizeof(__bf16));  // h pong, lo
  float* h_fp   = (float*)take(HS * sizeof(float));    // fp32 mirror -> h_n
  float* c_st   = (float*)take(HS * sizeof(float));

  // 1) one-time prep: weight transpose + split, h0 split, c0 copy
  wt_transpose<<<(unsigned)(WSZ / 256), 256, 0, stream>>>(W_i, W_h, Wti_h, Wti_l,
                                                          Wth_h, Wth_l);
  h0_convert<<<(unsigned)(HS / 256), 256, 0, stream>>>(h0, hbA_h, hbA_l);
  (void)hipMemcpyAsync(c_st, c0, HS * sizeof(float), hipMemcpyDeviceToDevice,
                       stream);

  for (int l = 0; l < LAYERS; ++l) {
    const float* Ain = (l == 0) ? x : h0_all;
    // parallel half: input-projection GEMM for the whole sequence
    gemm_gates<<<dim3(NG / 64, MTOT / 64), 256, 0, stream>>>(
        Ain, Wti_h + (size_t)l * NG * KDIM, Wti_l + (size_t)l * NG * KDIM, G,
        (l == 0) ? 1 : 0);

    __bf16* ph = hbA_h + (size_t)l * BB * HIDN;
    __bf16* pl = hbA_l + (size_t)l * BB * HIDN;
    __bf16* qh = hbB_h + (size_t)l * BB * HIDN;
    __bf16* ql = hbB_l + (size_t)l * BB * HIDN;
    float* cs = c_st + (size_t)l * BB * HIDN;
    float* hs = h_fp + (size_t)l * BB * HIDN;
    float* hout = (l == 0) ? h0_all : out;
    const float* bi_l = b_i + (size_t)l * NG;
    const float* bh_l = b_h + (size_t)l * NG;

    // serial half: recurrence (weights + h planes stay resident in 192MB L2)
    for (int t = 0; t < LL; ++t) {
      lstm_step<<<HIDN / 32, 256, 0, stream>>>(
          G, Wth_h + (size_t)l * NG * KDIM, Wth_l + (size_t)l * NG * KDIM,
          bi_l, bh_l, ph, pl, qh, ql, hs, cs, hout, t, (l == 0) ? 0 : 1);
      __bf16* th = ph; ph = qh; qh = th;
      __bf16* tl = pl; pl = ql; ql = tl;
    }
  }

  // outputs: [output | h_n | c_n]
  const size_t OUT0 = (size_t)BB * LL * HIDN;  // 33,554,432
  (void)hipMemcpyAsync(out + OUT0, h_fp, HS * sizeof(float),
                       hipMemcpyDeviceToDevice, stream);
  (void)hipMemcpyAsync(out + OUT0 + HS, c_st, HS * sizeof(float),
                       hipMemcpyDeviceToDevice, stream);
}
